// BiRNN_27358941676083
// MI455X (gfx1250) — matrix operand order repaired and checked by reading, not compiled, not run on hardware
//
#include <hip/hip_runtime.h>
#include <hip/hip_bf16.h>
#include <math.h>

#define N_ 4
#define T_ 1024
#define V_ 32000
#define H_ 512

typedef __attribute__((ext_vector_type(16))) __bf16 v16bf;
typedef __attribute__((ext_vector_type(8)))  float  v8f;
typedef __attribute__((ext_vector_type(4)))  int    v4i;
typedef unsigned short u16;
typedef unsigned int   u32;

#if __has_builtin(__builtin_amdgcn_global_load_async_to_lds_b128) && \
    __has_builtin(__builtin_amdgcn_s_wait_asynccnt)
#define USE_ASYNC_LDS 1
#else
#define USE_ASYNC_LDS 0
#endif

// ---- bf16 helpers (round-to-nearest-even), union-based for portability ----
__device__ __forceinline__ u16 f32_to_bf16_bits(float f) {
    union { float f; u32 u; } c; c.f = f;
    u32 r = c.u + 0x7FFFu + ((c.u >> 16) & 1u);
    return (u16)(r >> 16);
}
__device__ __forceinline__ float bf16_bits_to_f32(u16 b) {
    union { u32 u; float f; } c; c.u = ((u32)b) << 16;
    return c.f;
}

// A-matrix 16x32 bf16 fragment: element i of a lane maps to K =
// i + (lane>=16 ? 16 : 0)   (contiguous order, matches packed B)
__device__ __forceinline__ int a_k_of(int i, int lane) {
    return i + ((lane >> 4) << 4);
}

union BFrag { v16bf v; u16 u[16]; u32 w[8]; };

// ---------------------------------------------------------------------------
// Pack an f32 row-major [K x Ncols] matrix into WMMA B-fragment-ordered bf16
// tiles: tile (jt,kt) -> 32 lanes x 16 bf16 contiguous (32B per lane).
// B element i of lane L := W[kt*32 + i + ((L>>4)<<4)][jt*16 + (L&15)]
// ---------------------------------------------------------------------------
__global__ void pack_b_frags(const float* __restrict__ W, u16* __restrict__ P,
                             int K, int Ncols) {
    int KT = K >> 5;
    long long tid = (long long)blockIdx.x * blockDim.x + threadIdx.x;
    long long total = (long long)K * Ncols;
    if (tid >= total) return;
    int i = (int)(tid & 15);
    int L = (int)((tid >> 4) & 31);
    long long tile = tid >> 9;
    int kt = (int)(tile % KT);
    int jt = (int)(tile / KT);
    int k = kt * 32 + i + ((L >> 4) << 4);
    int j = jt * 16 + (L & 15);
    P[tid] = f32_to_bf16_bits(W[(long long)k * Ncols + j]);
}

// ---------------------------------------------------------------------------
// Bidirectional RNN scan. grid=2 (one block per direction), block=512 (16 waves).
// h state (4x512, padded to 16x512) double-buffered in LDS as bf16.
// ---------------------------------------------------------------------------
__global__ __launch_bounds__(512)
void birnn_scan(const int* __restrict__ x,
                const float* __restrict__ embed_f, const u16* __restrict__ WhP_f,
                const float* __restrict__ bh_f,
                const float* __restrict__ embed_b, const u16* __restrict__ WhP_b,
                const float* __restrict__ bh_b,
                u16* __restrict__ z,          // [N*T][2H] bf16 row-major
                float* __restrict__ hout)     // [2][N][H] f32
{
    const int dir = blockIdx.x;
    const float* embed = dir ? embed_b : embed_f;
    const u16*   WhP   = dir ? WhP_b   : WhP_f;
    const float* bh    = dir ? bh_b    : bh_f;
    float* hdst = hout + (size_t)dir * (N_ * H_);

    __shared__ __align__(16) u16 hA[2][16 * H_];   // 32 KB

    const int lane = threadIdx.x & 31;
    const int wave = threadIdx.x >> 5;
    const int jt0  = wave * 2;

    for (int e = threadIdx.x; e < 16 * H_; e += blockDim.x) {
        hA[0][e] = 0; hA[1][e] = 0;
    }
    __syncthreads();

    for (int it = 0; it < T_; ++it) {
        const int t  = dir ? (T_ - 1 - it) : it;
        const int pb = it & 1;          // read buffer (h_{t-1})
        const int wb = pb ^ 1;          // write buffer (h_t)
        __syncthreads();

        v8f acc0 = {}; v8f acc1 = {};
        const int m = lane & 15;
        #pragma unroll 4
        for (int kt = 0; kt < 16; ++kt) {
            const int kb = kt * 32;
            BFrag af;
            #pragma unroll
            for (int ii = 0; ii < 8; ++ii) {       // paired (even K) dword loads
                int k = a_k_of(2 * ii, lane);
                af.w[ii] = *reinterpret_cast<const u32*>(&hA[pb][m * H_ + kb + k]);
            }
            const v16bf b0 = *reinterpret_cast<const v16bf*>(
                WhP + (((size_t)jt0 * 16 + kt) * 32 + lane) * 16);
            const v16bf b1 = *reinterpret_cast<const v16bf*>(
                WhP + (((size_t)(jt0 + 1) * 16 + kt) * 32 + lane) * 16);
            acc0 = __builtin_amdgcn_wmma_f32_16x16x32_bf16(
                false, af.v, false, b0, (short)0, acc0, false, false);
            acc1 = __builtin_amdgcn_wmma_f32_16x16x32_bf16(
                false, af.v, false, b1, (short)0, acc1, false, false);
        }

        // Epilogue: lanes 0-15 hold rows M=0..7 in acc[r]; only rows 0..3 valid.
        if (lane < 16) {
            #pragma unroll
            for (int r = 0; r < N_; ++r) {
                const int n    = r;
                const int idx  = x[n * T_ + t];
                const int col0 = jt0 * 16 + lane;
                const int col1 = col0 + 16;
                float v0 = tanhf(acc0[r] + embed[(size_t)idx * H_ + col0] + bh[col0]);
                float v1 = tanhf(acc1[r] + embed[(size_t)idx * H_ + col1] + bh[col1]);
                u16 b0 = f32_to_bf16_bits(v0);
                u16 b1 = f32_to_bf16_bits(v1);
                hA[wb][n * H_ + col0] = b0;
                hA[wb][n * H_ + col1] = b1;
                size_t zr = ((size_t)n * T_ + t) * (2 * H_) + (size_t)dir * H_;
                z[zr + col0] = b0;
                z[zr + col1] = b1;
            }
        }
    }
    __syncthreads();
    const int fb = ((T_ - 1) & 1) ^ 1;   // buffer holding h_{T-1}
    for (int e = threadIdx.x; e < N_ * H_; e += blockDim.x)
        hdst[e] = bf16_bits_to_f32(hA[fb][e]);
}

// ---------------------------------------------------------------------------
// Logits GEMM: y[4096 x 32000] = z[4096 x 1024] @ Wout + bout  (f32 out).
// block = 256 (8 waves). Workgroup tile: 64 rows x 128 cols; each wave owns
// one 16-col jtile and 4 M-tiles (B-fragment reused x4). The 64x1024 A-panel
// is staged through LDS in two 64KB halves via async global->LDS copies.
// grid = (2000/8, 4096/64) = (250, 64). K loop: 2 phases x 16 chunks of 32.
// ---------------------------------------------------------------------------
__global__ __launch_bounds__(256)
void logits_gemm(const u16* __restrict__ z, const u16* __restrict__ WoutP,
                 const float* __restrict__ bout, float* __restrict__ y)
{
    __shared__ __align__(16) u16 zA[64 * 512];      // 64 KB: 64 rows x 512 cols

    const int lane = threadIdx.x & 31;
    const int wave = threadIdx.x >> 5;
    const int rowbase = blockIdx.y * 64;
    const int jt = blockIdx.x * 8 + wave;            // 0..1999
    const int m  = lane & 15;

    v8f acc[4] = {{}, {}, {}, {}};

    for (int ph = 0; ph < 2; ++ph) {
        __syncthreads();                             // previous phase reads done
        // Stage A-panel half: rows [rowbase..+63], cols [ph*512..+511].
        // 4096 chunks of 16B; 256 threads x 16 chunks.
        for (int c = threadIdx.x; c < 4096; c += 256) {
            const int row = c >> 6;                  // 64 x 16B chunks per row
            const int cc  = (c & 63) * 8;            // u16 column within half
            const u16* gsrc = z + (size_t)(rowbase + row) * 1024 + ph * 512 + cc;
            u16* ldst = &zA[row * 512 + cc];
#if USE_ASYNC_LDS
            __builtin_amdgcn_global_load_async_to_lds_b128((v4i*)gsrc, (v4i*)ldst, 0, 0);
#else
            *reinterpret_cast<uint4*>(ldst) = *reinterpret_cast<const uint4*>(gsrc);
#endif
        }
#if USE_ASYNC_LDS
        __builtin_amdgcn_s_wait_asynccnt(0);
#endif
        __syncthreads();

        #pragma unroll 2
        for (int kt = 0; kt < 16; ++kt) {
            const int kb = kt * 32;
            const int ktg = ph * 16 + kt;            // global K-chunk 0..31
            const v16bf b = *reinterpret_cast<const v16bf*>(
                WoutP + (((size_t)jt * 32 + ktg) * 32 + lane) * 16);
            #pragma unroll
            for (int mt = 0; mt < 4; ++mt) {
                BFrag af;
                #pragma unroll
                for (int ii = 0; ii < 8; ++ii) {
                    int k = a_k_of(2 * ii, lane);
                    af.w[ii] = *reinterpret_cast<const u32*>(
                        &zA[(mt * 16 + m) * 512 + kb + k]);
                }
                acc[mt] = __builtin_amdgcn_wmma_f32_16x16x32_bf16(
                    false, af.v, false, b, (short)0, acc[mt], false, false);
            }
        }
    }

    const int col = jt * 16 + m;
    const float bo = bout[col];
    const int mhi = (lane >> 4) << 3;
    #pragma unroll
    for (int mt = 0; mt < 4; ++mt) {
        #pragma unroll
        for (int r = 0; r < 8; ++r) {
            const int row = rowbase + mt * 16 + r + mhi;
            y[(size_t)row * V_ + col] = acc[mt][r] + bo;
        }
    }
}

// ---------------------------------------------------------------------------
// Row softmax over y[4096 x 32000], in place. One block per row.
// ---------------------------------------------------------------------------
__global__ __launch_bounds__(256)
void softmax_rows(float* __restrict__ y)
{
    __shared__ float red[256];
    float* p = y + (size_t)blockIdx.x * V_;
    const int tid = threadIdx.x;

    float mx = -3.402823466e38f;
    for (int c = tid; c < V_; c += 256) mx = fmaxf(mx, p[c]);
    red[tid] = mx; __syncthreads();
    for (int s = 128; s > 0; s >>= 1) {
        if (tid < s) red[tid] = fmaxf(red[tid], red[tid + s]);
        __syncthreads();
    }
    mx = red[0]; __syncthreads();

    float sum = 0.f;
    for (int c = tid; c < V_; c += 256) sum += __expf(p[c] - mx);
    red[tid] = sum; __syncthreads();
    for (int s = 128; s > 0; s >>= 1) {
        if (tid < s) red[tid] += red[tid + s];
        __syncthreads();
    }
    const float inv = 1.0f / red[0];
    for (int c = tid; c < V_; c += 256) p[c] = __expf(p[c] - mx) * inv;
}

// ---------------------------------------------------------------------------
extern "C" void kernel_launch(void* const* d_in, const int* in_sizes, int n_in,
                              void* d_out, int out_size, void* d_ws, size_t ws_size,
                              hipStream_t stream) {
    (void)in_sizes; (void)n_in; (void)out_size; (void)ws_size;
    const int*   x       = (const int*)  d_in[0];
    const float* embed_f = (const float*)d_in[1];
    const float* Wh_f    = (const float*)d_in[2];
    const float* bh_f    = (const float*)d_in[3];
    const float* embed_b = (const float*)d_in[4];
    const float* Wh_b    = (const float*)d_in[5];
    const float* bh_b    = (const float*)d_in[6];
    const float* Wout    = (const float*)d_in[7];
    const float* bout    = (const float*)d_in[8];

    float* y    = (float*)d_out;                       // [N,T,V]
    float* hout = y + (size_t)N_ * T_ * V_;            // [2,N,H]

    u16* WhP_f = (u16*)d_ws;                           // 512*512 bf16
    u16* WhP_b = WhP_f + 512 * 512;                    // 512*512 bf16
    u16* WoutP = WhP_b + 512 * 512;                    // 1024*32000 bf16
    u16* zbuf  = WoutP + (size_t)1024 * V_;            // 4096*1024 bf16

    // 1) pack weights into WMMA B-fragment bf16 layout
    pack_b_frags<<<(512 * 512 + 255) / 256, 256, 0, stream>>>(Wh_f, WhP_f, 512, 512);
    pack_b_frags<<<(512 * 512 + 255) / 256, 256, 0, stream>>>(Wh_b, WhP_b, 512, 512);
    {
        long long tot = (long long)1024 * V_;
        pack_b_frags<<<(unsigned)((tot + 255) / 256), 256, 0, stream>>>(Wout, WoutP, 1024, V_);
    }

    // 2) bidirectional recurrent scan (both directions concurrently)
    birnn_scan<<<2, 512, 0, stream>>>(x, embed_f, WhP_f, bh_f,
                                      embed_b, WhP_b, bh_b, zbuf, hout);

    // 3) big WMMA GEMM -> logits (f32) into d_out
    logits_gemm<<<dim3(250, 64), 256, 0, stream>>>(zbuf, WoutP, bout, y);

    // 4) row softmax in place
    softmax_rows<<<N_ * T_, 256, 0, stream>>>(y);
}
